// KDALayer_62302795596517
// MI455X (gfx1250) — matrix operand order repaired and checked by reading, not compiled, not run on hardware
//
#include <hip/hip_runtime.h>
#include <cstdint>
#include <cstddef>

// Problem constants (match reference)
#define BD   4
#define TD   2048
#define DMD  1024
#define HD   16
#define DHD  64
#define DID  1024
#define MTOT (BD * TD)   // 8192 rows
#define PSTR 5152        // fused projection width: 5*1024 (q,k,v,a,aR) + 16 (b) + 16 (g)

typedef __attribute__((ext_vector_type(8)))  __bf16 v8bf;
typedef __attribute__((ext_vector_type(16))) __bf16 v16bf;
typedef __attribute__((ext_vector_type(8)))  float  v8f;
typedef int vi4 __attribute__((vector_size(16)));   // pointee type for async b128 builtin

#define GAS __attribute__((address_space(1)))
#define LAS __attribute__((address_space(3)))

#if defined(__has_builtin)
#if __has_builtin(__builtin_amdgcn_global_load_async_to_lds_b128) && \
    __has_builtin(__builtin_amdgcn_global_load_async_to_lds_b32) &&  \
    __has_builtin(__builtin_amdgcn_s_wait_asynccnt)
#define HAVE_ASYNC_LDS 1
#endif
#endif

// ---------------------------------------------------------------------------
// f32 -> bf16 conversion (weights)
// ---------------------------------------------------------------------------
__global__ void cvt_f32_bf16(const float* __restrict__ in, __bf16* __restrict__ out, int n) {
    int idx = blockIdx.x * blockDim.x + threadIdx.x;
    int stride = gridDim.x * blockDim.x;
    for (; idx < n; idx += stride) out[idx] = (__bf16)in[idx];
}

// ---------------------------------------------------------------------------
// Fused LayerNorm -> bf16 activation panel. One block per row, 256 threads.
// ---------------------------------------------------------------------------
__global__ __launch_bounds__(256)
void ln_bf16(const float* __restrict__ x, const float* __restrict__ w,
             const float* __restrict__ b, __bf16* __restrict__ xn) {
    __shared__ float s1[256], s2[256];
    const int row = blockIdx.x, tid = threadIdx.x;
    const float* xr = x + (size_t)row * DMD;
    float4 xv = reinterpret_cast<const float4*>(xr)[tid];
    s1[tid] = xv.x + xv.y + xv.z + xv.w;
    s2[tid] = xv.x * xv.x + xv.y * xv.y + xv.z * xv.z + xv.w * xv.w;
    __syncthreads();
    for (int s = 128; s > 0; s >>= 1) {
        if (tid < s) { s1[tid] += s1[tid + s]; s2[tid] += s2[tid + s]; }
        __syncthreads();
    }
    const float mu  = s1[0] * (1.0f / DMD);
    const float var = s2[0] * (1.0f / DMD) - mu * mu;
    const float inv = rsqrtf(var + 1e-5f);
    float4 wv = reinterpret_cast<const float4*>(w)[tid];
    float4 bv = reinterpret_cast<const float4*>(b)[tid];
    __bf16* o = xn + (size_t)row * DMD + tid * 4;
    o[0] = (__bf16)((xv.x - mu) * inv * wv.x + bv.x);
    o[1] = (__bf16)((xv.y - mu) * inv * wv.y + bv.y);
    o[2] = (__bf16)((xv.z - mu) * inv * wv.z + bv.z);
    o[3] = (__bf16)((xv.w - mu) * inv * wv.w + bv.w);
}

// ---------------------------------------------------------------------------
// bf16 WMMA GEMM:  C[M,N] = A[M,K] * B[N,K]^T  (+ optional residual)
// One wave computes a 32(M) x 64(N) macro-tile: 2 A-fragments x 4 B-fragments
// -> 8 v_wmma per k-step against 6 fragment loads (A reused 4x, B reused 2x).
// A/B fragments use contiguous k = 16*(lane/16) + s; C rows 8*(lane/16) + r (wave32).
// ---------------------------------------------------------------------------
__global__ __launch_bounds__(256)
void wmma_gemm_bf16(const __bf16* __restrict__ A, const __bf16* __restrict__ Bm,
                    float* __restrict__ C, const float* __restrict__ resid,
                    int M, int N, int K, int tilesN) {
    const int wave = (blockIdx.x * 256 + threadIdx.x) >> 5;
    const int mt = wave / tilesN;   // 32-row macro tile index
    const int nb = wave % tilesN;   // 64-col macro tile index
    if (mt * 32 >= M) return;       // uniform per wave -> EXEC stays all-ones
    const int lane = threadIdx.x & 31;
    const int hi   = lane >> 4;
    const int l15  = lane & 15;

    const __bf16* arow0 = A + (size_t)(mt * 32 + l15) * K + hi * 16;
    const __bf16* arow1 = arow0 + (size_t)16 * K;
    const __bf16* brow[4];
#pragma unroll
    for (int t = 0; t < 4; ++t) {
        int rB = nb * 64 + t * 16 + l15;
        rB = rB < N ? rB : N - 1;   // clamp OOB tiles (results discarded)
        brow[t] = Bm + (size_t)rB * K + hi * 16;
    }

    v8f acc[8] = {};
    for (int kk = 0; kk < K; kk += 32) {
        v8bf a00 = *reinterpret_cast<const v8bf*>(arow0 + kk);
        v8bf a01 = *reinterpret_cast<const v8bf*>(arow0 + kk + 8);
        v8bf a10 = *reinterpret_cast<const v8bf*>(arow1 + kk);
        v8bf a11 = *reinterpret_cast<const v8bf*>(arow1 + kk + 8);
        v16bf af0 = __builtin_shufflevector(a00, a01, 0, 1, 2, 3, 4, 5, 6, 7,
                                                      8, 9, 10, 11, 12, 13, 14, 15);
        v16bf af1 = __builtin_shufflevector(a10, a11, 0, 1, 2, 3, 4, 5, 6, 7,
                                                      8, 9, 10, 11, 12, 13, 14, 15);
        __builtin_prefetch(arow0 + kk + 128, 0, 0);
#pragma unroll
        for (int t = 0; t < 4; ++t) {
            v16bf bfr = *reinterpret_cast<const v16bf*>(brow[t] + kk);
            acc[t]     = __builtin_amdgcn_wmma_f32_16x16x32_bf16(
                false, af0, false, bfr, (short)0, acc[t], false, false);
            acc[4 + t] = __builtin_amdgcn_wmma_f32_16x16x32_bf16(
                false, af1, false, bfr, (short)0, acc[4 + t], false, false);
        }
    }

#pragma unroll
    for (int t = 0; t < 4; ++t) {
        const int col = nb * 64 + t * 16 + l15;
        if (col < N) {
#pragma unroll
            for (int r = 0; r < 8; ++r) {
                const int row0 = mt * 32 + hi * 8 + r;
                const int row1 = row0 + 16;
                float v0 = acc[t][r], v1 = acc[4 + t][r];
                if (resid) {
                    v0 += resid[(size_t)row0 * N + col];
                    v1 += resid[(size_t)row1 * N + col];
                }
                C[(size_t)row0 * N + col] = v0;
                C[(size_t)row1 * N + col] = v1;
            }
        }
    }
}

// ---------------------------------------------------------------------------
// Sequential delta-rule scan. One workgroup per (b,h); both 64x64 states in
// LDS (element (j,i) at j*64+i: thread i owns column i, conflict-free).
// 256 threads = 4-way split of the j-reduction. Next step's inputs are
// DMA-prefetched into an LDS double buffer with GLOBAL_LOAD_ASYNC_TO_LDS
// (ASYNCcnt) so global latency overlaps the recurrence.
// ---------------------------------------------------------------------------
__global__ __launch_bounds__(256)
void kda_scan(const float* __restrict__ proj,  // (MTOT, PSTR) fused q|k|v|a|aR|b|g
              const float* __restrict__ ba, const float* __restrict__ baR,
              const float* __restrict__ bbb, const float* __restrict__ bgg,
              __bf16* __restrict__ oo) {
    __shared__ float S[4096];
    __shared__ float Rs[4096];
    __shared__ float kn[64], av[64], aRv[64], qv[64], vv[64], uarr[64], rarr[64];
    __shared__ float p0[256], p1[256], p2[256];
    __shared__ float msc[2];                      // beta, gamma (sigmoided)
    __shared__ __align__(16) float pref[2][336];  // 5x64 inputs + 2 scalars

    const int tid = threadIdx.x;
    const int i   = tid & 63;
    const int jb  = tid >> 6;
    const int jlo = jb * 16;
    const int b   = blockIdx.x >> 4;
    const int h   = blockIdx.x & 15;
    const int ch  = h * 64 + i;

    for (int idx = tid; idx < 4096; idx += 256) { S[idx] = 0.0f; Rs[idx] = 0.0f; }
    const float bai = ba[ch], baRi = baR[ch];

#if defined(HAVE_ASYNC_LDS)
    auto prefetch = [&](int t, float* buf) {
        if (tid < 80) {                       // 5 arrays x 16 lanes x 16B
            const int a = tid >> 4, e = tid & 15;
            const float* gp = proj + ((size_t)b * TD + t) * PSTR + a * 1024 + h * 64 + e * 4;
            float* lp = buf + a * 64 + e * 4;
            __builtin_amdgcn_global_load_async_to_lds_b128(
                (GAS vi4*)(uintptr_t)gp, (LAS vi4*)(uint32_t)(uintptr_t)lp, 0, 0);
        } else if (tid < 82) {                // beta / gamma raw scalars
            const float* gp = proj + ((size_t)b * TD + t) * PSTR + 5120 + (tid == 81 ? 16 : 0) + h;
            float* lp = buf + 320 + (tid - 80);
            __builtin_amdgcn_global_load_async_to_lds_b32(
                (GAS int*)(uintptr_t)gp, (LAS int*)(uint32_t)(uintptr_t)lp, 0, 0);
        }
    };
    prefetch(0, pref[0]);
#endif
    __syncthreads();

    for (int t = 0; t < TD; ++t) {
        const size_t rowb = (size_t)b * TD + t;
#if defined(HAVE_ASYNC_LDS)
        __builtin_amdgcn_s_wait_asynccnt(0);   // pref[t&1] landed (no-op for non-issuers)
        __syncthreads();                        // make DMA-written LDS visible to all waves
        const float* cb = pref[t & 1];
        if (t + 1 < TD) prefetch(t + 1, pref[(t + 1) & 1]);
        if (jb == 0) {
            qv[i] = cb[i];
            kn[i] = cb[64 + i];                // raw k staged here
            vv[i] = cb[128 + i];
            av[i]  = 1.0f / (1.0f + __expf(-(cb[192 + i] + bai)));
            aRv[i] = 1.0f / (1.0f + __expf(-(cb[256 + i] + baRi)));
        } else if (tid == 64) {
            msc[0] = 1.0f / (1.0f + __expf(-(cb[320] + bbb[h])));
            msc[1] = 1.0f / (1.0f + __expf(-(cb[321] + bgg[h])));
        }
#else
        if (jb == 0) {
            const size_t g = rowb * PSTR + ch;
            qv[i] = proj[g];
            kn[i] = proj[g + 1024];
            vv[i] = proj[g + 2048];
            av[i]  = 1.0f / (1.0f + __expf(-(proj[g + 3072] + bai)));
            aRv[i] = 1.0f / (1.0f + __expf(-(proj[g + 4096] + baRi)));
        } else if (tid == 64) {
            msc[0] = 1.0f / (1.0f + __expf(-(proj[rowb * PSTR + 5120 + h] + bbb[h])));
            msc[1] = 1.0f / (1.0f + __expf(-(proj[rowb * PSTR + 5136 + h] + bgg[h])));
        }
#endif
        __syncthreads();
        // k normalization (broadcast LDS reads; stage to avoid RAW on kn)
        if (jb == 0) {
            float ss = 0.0f;
#pragma unroll 8
            for (int j = 0; j < 64; ++j) { float kj = kn[j]; ss += kj * kj; }
            float dn = sqrtf(ss);
            dn = dn > 1e-12f ? dn : 1e-12f;
            p0[i] = kn[i] / dn;
        }
        __syncthreads();
        if (jb == 0) kn[i] = p0[i];
        __syncthreads();
        // pass 1: pred = S_old^T kn ; kp = (a.*S_old)^T kn ; kpR = (aR.*R_old)^T kn
        float pr = 0.0f, kp = 0.0f, kpR = 0.0f;
#pragma unroll 4
        for (int jj = 0; jj < 16; ++jj) {
            const int j = jlo + jj;
            const float knj = kn[j];
            const float s  = S[j * 64 + i];
            const float rr = Rs[j * 64 + i];
            pr  += knj * s;
            kp  += knj * av[j] * s;
            kpR += knj * aRv[j] * rr;
        }
        p0[tid] = pr; p1[tid] = kp; p2[tid] = kpR;
        __syncthreads();
        if (jb == 0) {
            const float pred = p0[i] + p0[64 + i] + p0[128 + i] + p0[192 + i];
            const float kpf  = p1[i] + p1[64 + i] + p1[128 + i] + p1[192 + i];
            const float kpRf = p2[i] + p2[64 + i] + p2[128 + i] + p2[192 + i];
            const float vi = vv[i];
            float r = vi - pred;
            r = r > 1.0f ? 1.0f : (r < -1.0f ? -1.0f : r);
            uarr[i] = msc[0] * (vi - kpf);    // beta  * (v - kn.(a.*S))
            rarr[i] = msc[1] * (r - kpRf);    // gamma * (r - kn.(aR.*R))
        }
        __syncthreads();
        // state update + output accumulation (o = S_new^T q + R_new^T q)
        float o1p = 0.0f, o2p = 0.0f;
#pragma unroll 4
        for (int jj = 0; jj < 16; ++jj) {
            const int j = jlo + jj;
            const int idx = j * 64 + i;
            const float knj = kn[j], qj = qv[j];
            const float s = av[j] * S[idx] + knj * uarr[i];
            S[idx] = s;  o1p += s * qj;
            const float rr = aRv[j] * Rs[idx] + knj * rarr[i];
            Rs[idx] = rr; o2p += rr * qj;
        }
        p0[tid] = o1p; p1[tid] = o2p;
        __syncthreads();
        if (jb == 0) {
            const float oa = p0[i] + p0[64 + i] + p0[128 + i] + p0[192 + i]
                           + p1[i] + p1[64 + i] + p1[128 + i] + p1[192 + i];
            oo[rowb * DID + ch] = (__bf16)oa;
        }
        __syncthreads();
    }
}

// ---------------------------------------------------------------------------
// Host-side orchestration
// ---------------------------------------------------------------------------
extern "C" void kernel_launch(void* const* d_in, const int* in_sizes, int n_in,
                              void* d_out, int out_size, void* d_ws, size_t ws_size,
                              hipStream_t stream) {
    (void)in_sizes; (void)n_in; (void)out_size; (void)ws_size;
    const float* x    = (const float*)d_in[0];
    const float* Wq   = (const float*)d_in[1];
    const float* Wk   = (const float*)d_in[2];
    const float* Wv   = (const float*)d_in[3];
    const float* Wa   = (const float*)d_in[4];
    const float* ba   = (const float*)d_in[5];
    const float* Wb   = (const float*)d_in[6];
    const float* bb   = (const float*)d_in[7];
    const float* Wg   = (const float*)d_in[8];
    const float* bg   = (const float*)d_in[9];
    const float* WaR  = (const float*)d_in[10];
    const float* baR  = (const float*)d_in[11];
    const float* Wo   = (const float*)d_in[12];
    const float* ln_w = (const float*)d_in[13];
    const float* ln_b = (const float*)d_in[14];
    float* out = (float*)d_out;

    // Workspace carve (all offsets 256B aligned)
    char* p = (char*)d_ws;
    auto carve = [&](size_t bytes) -> char* {
        char* r = p;
        p += (bytes + 255) & ~(size_t)255;
        return r;
    };
    __bf16* xn   = (__bf16*)carve((size_t)MTOT * DMD * 2);
    __bf16* Wcat = (__bf16*)carve((size_t)PSTR * DMD * 2);  // stacked q|k|v|a|aR|b|g
    __bf16* Wo_b = (__bf16*)carve((size_t)DMD * DID * 2);
    float*  projf = (float*)carve((size_t)MTOT * PSTR * 4);
    __bf16* o_b  = (__bf16*)carve((size_t)MTOT * DID * 2);

    // 1. Convert + stack weights to bf16 (rows of Wcat: [Wq;Wk;Wv;Wa;WaR;Wb;Wg])
    const int nW = DID * DMD;
    cvt_f32_bf16<<<1024, 256, 0, stream>>>(Wq,  Wcat + (size_t)0 * nW, nW);
    cvt_f32_bf16<<<1024, 256, 0, stream>>>(Wk,  Wcat + (size_t)1 * nW, nW);
    cvt_f32_bf16<<<1024, 256, 0, stream>>>(Wv,  Wcat + (size_t)2 * nW, nW);
    cvt_f32_bf16<<<1024, 256, 0, stream>>>(Wa,  Wcat + (size_t)3 * nW, nW);
    cvt_f32_bf16<<<1024, 256, 0, stream>>>(WaR, Wcat + (size_t)4 * nW, nW);
    cvt_f32_bf16<<<64,  256, 0, stream>>>(Wb,  Wcat + (size_t)5 * nW, HD * DMD);
    cvt_f32_bf16<<<64,  256, 0, stream>>>(Wg,  Wcat + (size_t)5 * nW + (size_t)HD * DMD, HD * DMD);
    cvt_f32_bf16<<<1024, 256, 0, stream>>>(Wo, Wo_b, nW);

    // 2. LayerNorm -> bf16
    ln_bf16<<<MTOT, 256, 0, stream>>>(x, ln_w, ln_b, xn);

    // 3. All projections in ONE fused WMMA GEMM (M=8192, N=5152, K=1024)
    auto gemm = [&](const __bf16* A, const __bf16* Bm, float* C,
                    const float* resid, int M, int N, int K) {
        const int tilesN = (N + 63) / 64;
        const int mtiles = (M + 31) / 32;
        const int waves = mtiles * tilesN;
        const int blocks = (waves + 7) / 8;
        wmma_gemm_bf16<<<blocks, 256, 0, stream>>>(A, Bm, C, resid, M, N, K, tilesN);
    };
    gemm(xn, Wcat, projf, nullptr, MTOT, PSTR, DMD);

    // 4. Sequential scan: one block per (b,h), async-LDS double-buffered inputs
    kda_scan<<<BD * HD, 256, 0, stream>>>(projf, ba, baR, bb, bg, o_b);

    // 5. Output projection + residual
    gemm(o_b, Wo_b, out, x, MTOT, DMD, DID);
}
